// ResNet_24464133718187
// MI455X (gfx1250) — 8-bit matrix operand order repaired and checked by reading, compiled, not run on hardware
//
#include <hip/hip_runtime.h>
#include <hip/hip_bf16.h>

typedef signed char s8;
typedef __attribute__((ext_vector_type(8))) int v8i;

__device__ __forceinline__ v8i v8i_zero() {
    v8i v = {0, 0, 0, 0, 0, 0, 0, 0};
    return v;
}

__device__ __forceinline__ constexpr int kbaseA(int d, int half) {
    return ((d >> 2) << 5) + (((d >> 1) & 1) << 4) + ((d & 1) << 2) + (half << 3);
}
__device__ __forceinline__ constexpr int kbaseB(int d, int half) {
    return ((d >> 2) << 5) + (((d >> 1) & 1) << 4) + ((d & 1) << 2) + (half << 3);
}

__global__ void k_zero8(s8* __restrict__ p, int n) {
    int i = blockIdx.x * 256 + threadIdx.x;
    if (i < n) p[i] = 0;
}

__global__ void k_binw(const float* __restrict__ w, s8* __restrict__ wq,
                       int Cout, int Cin, int T) {
    int idx = blockIdx.x * 256 + threadIdx.x;
    int total = Cout * Cin * T;
    if (idx >= total) return;
    int i = idx % Cin;
    int o = (idx / Cin) % Cout;
    int t = idx / (Cin * Cout);
    float v = w[(size_t)(o * Cin + i) * T + t];
    wq[idx] = (s8)((v > 0.f) ? 1 : ((v < 0.f) ? -1 : 0));
}

__global__ void k_conv1(const float* __restrict__ x, const float* __restrict__ w,
                        float* __restrict__ out, int N) {
    __shared__ float ws[864];
    int t = threadIdx.x;
    for (int j = t; j < 864; j += 256) {
        float v = w[j];
        ws[j] = (v > 0.f) ? 1.f : ((v < 0.f) ? -1.f : 0.f);
    }
    __syncthreads();
    long i = (long)blockIdx.x * 256 + t;
    long npix = (long)N * 32 * 32;
    if (i >= npix) return;
    int xx = (int)(i & 31);
    int yy = (int)((i >> 5) & 31);
    int n = (int)(i >> 10);
    float acc[32];
#pragma unroll
    for (int co = 0; co < 32; ++co) acc[co] = 0.f;
    for (int ky = 0; ky < 3; ++ky) {
        int iy = yy + ky - 1;
        if (iy < 0 || iy >= 32) continue;
        for (int kx = 0; kx < 3; ++kx) {
            int ix = xx + kx - 1;
            if (ix < 0 || ix >= 32) continue;
#pragma unroll
            for (int ci = 0; ci < 3; ++ci) {
                float v = x[((long)(n * 3 + ci) * 32 + iy) * 32 + ix];
#pragma unroll
                for (int co = 0; co < 32; ++co)
                    acc[co] = fmaf(v, ws[((co * 3 + ci) * 3 + ky) * 3 + kx], acc[co]);
            }
        }
    }
#pragma unroll
    for (int co = 0; co < 32; ++co) out[i * 32 + co] = acc[co];
}

template <int CIN, int NTILES, int KSZ, int STRIDE, int PAD>
__global__ __launch_bounds__(32) void k_bconv(const s8* __restrict__ qin,
                                              const s8* __restrict__ wq,
                                              float* __restrict__ out,
                                              int N, int H, int W, int Hout, int Wout) {
    constexpr int LW = 15 * STRIDE + KSZ;
    constexpr int NDW = KSZ * LW * CIN / 4;
    constexpr int CW = CIN / 4;
    __shared__ s8 smem[KSZ * LW * CIN];

    int lane = threadIdx.x;
    int m = lane & 15, half = lane >> 4;
    int tilesPerRow = Wout >> 4;
    int tile = blockIdx.x;
    int oxbase = (tile % tilesPerRow) << 4;
    int rowid = tile / tilesPerRow;
    int oy = rowid % Hout;
    int n = rowid / Hout;

    int x0 = oxbase * STRIDE - PAD;
    int y0 = oy * STRIDE - PAD;
    for (int j = lane; j < NDW; j += 32) {
        int kbo = (j % CW) << 2;
        int xi = (j / CW) % LW;
        int ri = j / (CW * LW);
        int gx = x0 + xi;
        int gy = y0 + ri;
        int v = 0;
        if (gx >= 0 && gx < W && gy >= 0 && gy < H)
            v = *(const int*)(qin + ((long)(n * H + gy) * W + gx) * CIN + kbo);
        *(int*)(smem + (size_t)j * 4) = v;
    }
    __syncthreads();

    v8i acc[NTILES];
#pragma unroll
    for (int tI = 0; tI < NTILES; ++tI) acc[tI] = v8i_zero();

#pragma unroll
    for (int ky = 0; ky < KSZ; ++ky) {
#pragma unroll
        for (int kx = 0; kx < KSZ; ++kx) {
            const s8* arow = smem + ((size_t)(ky * LW) + (m * STRIDE + kx)) * CIN;
            v8i a;
#pragma unroll
            for (int d = 0; d < 8; ++d) {
                constexpr_if_helper:;
                int kb = kbaseA(d, half);
                if (kbaseA(d, 0) < CIN)
                    a[d] = *(const int*)(arow + kb);
                else
                    a[d] = 0;
            }
            const s8* wt = wq + (size_t)(ky * KSZ + kx) * (NTILES * 16) * CIN;
#pragma unroll
            for (int tI = 0; tI < NTILES; ++tI) {
                const s8* bb = wt + (size_t)((tI << 4) + m) * CIN;
                v8i b;
#pragma unroll
                for (int d = 0; d < 8; ++d) {
                    int kb = kbaseB(d, half);
                    if (kbaseB(d, 0) < CIN)
                        b[d] = *(const int*)(bb + kb);
                    else
                        b[d] = 0;
                }
                acc[tI] = __builtin_amdgcn_wmma_i32_16x16x64_iu8(
                    true, a, true, b, acc[tI], false, false);
            }
        }
    }
    int Cout = NTILES * 16;
#pragma unroll
    for (int tI = 0; tI < NTILES; ++tI) {
#pragma unroll
        for (int r = 0; r < 8; ++r) {
            int mo = r + (half << 3);
            out[((size_t)(n * Hout + oy) * Wout + (oxbase + mo)) * Cout + (tI << 4) + m] =
                (float)acc[tI][r];
        }
    }
}

#define STAT_CHUNK 65536L
__global__ void k_stats_part(const float* __restrict__ x, float* __restrict__ part,
                             long total, int C) {
    __shared__ float sh1[256], sh2[256];
    int t = threadIdx.x;
    long base = (long)blockIdx.x * STAT_CHUNK;
    long end = base + STAT_CHUNK;
    if (end > total) end = total;
    float s1 = 0.f, s2 = 0.f;
    for (long i = base + t; i < end; i += 256) {
        float v = x[i];
        s1 += v;
        s2 += v * v;
    }
    sh1[t] = s1;
    sh2[t] = s2;
    __syncthreads();
    if (t < C) {
        float a = 0.f, b = 0.f;
        for (int j = t; j < 256; j += C) { a += sh1[j]; b += sh2[j]; }
        part[(size_t)blockIdx.x * (2 * C) + t] = a;
        part[(size_t)blockIdx.x * (2 * C) + C + t] = b;
    }
}

__global__ void k_stats_final(const float* __restrict__ part, int nblk, int C, int Cvalid,
                              const float* __restrict__ g, const float* __restrict__ b,
                              float* __restrict__ scale, float* __restrict__ shift,
                              float invCnt) {
    int c = threadIdx.x;
    if (c >= C) return;
    float s1 = 0.f, s2 = 0.f;
    for (int i = 0; i < nblk; ++i) {
        s1 += part[(size_t)i * (2 * C) + c];
        s2 += part[(size_t)i * (2 * C) + C + c];
    }
    float m = s1 * invCnt;
    float v = s2 * invCnt - m * m;
    float inv = rsqrtf(v + 1e-5f);
    float gg = (c < Cvalid) ? g[c] : 1.f;
    float bb = (c < Cvalid) ? b[c] : 0.f;
    float sc = gg * inv;
    scale[c] = sc;
    shift[c] = bb - m * sc;
}

__global__ void k_bnact(const float* __restrict__ x, const float* __restrict__ scale,
                        const float* __restrict__ shift, float* __restrict__ fout,
                        s8* __restrict__ qout, long total, int cmask, int do_ht) {
    long i = (long)blockIdx.x * 256 + threadIdx.x;
    if (i >= total) return;
    int c = (int)(i & cmask);
    float y = x[i] * scale[c] + shift[c];
    if (do_ht) y = fminf(1.f, fmaxf(-1.f, y));
    if (fout) fout[i] = y;
    if (qout) qout[i] = (s8)((y > 0.f) ? 1 : ((y < 0.f) ? -1 : 0));
}

__global__ void k_add(float* __restrict__ a, const float* __restrict__ b, long total) {
    long i = (long)blockIdx.x * 256 + threadIdx.x;
    if (i >= total) return;
    a[i] += b[i];
}

__global__ void k_avgpool(const float* __restrict__ x, float* __restrict__ out, int N) {
    long i = (long)blockIdx.x * 256 + threadIdx.x;
    long total = (long)N * 256;
    if (i >= total) return;
    int feat = (int)(i & 255);
    int n = (int)(i >> 8);
    int c = feat >> 2;
    int py = (feat >> 1) & 1;
    int px = feat & 1;
    float s = 0.f;
    for (int dy = 0; dy < 8; ++dy)
        for (int dx = 0; dx < 8; ++dx)
            s += x[((size_t)(n * 16 + py * 8 + dy) * 16 + (px * 8 + dx)) * 64 + c];
    out[i] = s * (1.f / 64.f);
}

__global__ __launch_bounds__(32) void k_fc(const s8* __restrict__ q, const s8* __restrict__ wq,
                                           const float* __restrict__ bias,
                                           float* __restrict__ out) {
    int lane = threadIdx.x;
    int m = lane & 15, half = lane >> 4;
    int row0 = blockIdx.x << 4;
    v8i acc = v8i_zero();
    for (int kc = 0; kc < 4; ++kc) {
        v8i a, b;
        const s8* ab = q + (size_t)(row0 + m) * 256 + (kc << 6);
        const s8* bb = wq + (size_t)m * 256 + (kc << 6);
#pragma unroll
        for (int d = 0; d < 8; ++d) a[d] = *(const int*)(ab + kbaseA(d, half));
#pragma unroll
        for (int d = 0; d < 8; ++d) b[d] = *(const int*)(bb + kbaseB(d, half));
        acc = __builtin_amdgcn_wmma_i32_16x16x64_iu8(true, a, true, b, acc, false, false);
    }
    float bv = (m < 10) ? bias[m] : 0.f;
#pragma unroll
    for (int r = 0; r < 8; ++r) {
        int mo = r + (half << 3);
        out[(size_t)(row0 + mo) * 16 + m] = (float)acc[r] + bv;
    }
}

__global__ void k_head(const float* __restrict__ fcb, const float* __restrict__ scale,
                       const float* __restrict__ shift, float* __restrict__ out, int N) {
    int n = blockIdx.x * 256 + threadIdx.x;
    if (n >= N) return;
    float v[10];
    float mx = -3.4e38f;
#pragma unroll
    for (int j = 0; j < 10; ++j) {
        v[j] = fcb[(size_t)n * 16 + j] * scale[j] + shift[j];
        mx = fmaxf(mx, v[j]);
    }
    float s = 0.f;
#pragma unroll
    for (int j = 0; j < 10; ++j) s += expf(v[j] - mx);
    float l = logf(s);
#pragma unroll
    for (int j = 0; j < 10; ++j) out[(size_t)n * 10 + j] = v[j] - mx - l;
}

static inline int nblocks(long n, int b = 256) { return (int)((n + b - 1) / b); }

extern "C" void kernel_launch(void* const* d_in, const int* in_sizes, int n_in,
                              void* d_out, int out_size, void* d_ws, size_t ws_size,
                              hipStream_t stream) {
    (void)in_sizes; (void)n_in; (void)out_size; (void)ws_size;
    const float* X = (const float*)d_in[0];
    auto F = [&](int i) { return (const float*)d_in[i]; };

    char* ws = (char*)d_ws;
    float* bufA = (float*)(ws);
    float* bufB = (float*)(ws + ((size_t)128 << 20));
    float* bufC = (float*)(ws + ((size_t)256 << 20));
    s8* qA = (s8*)(ws + ((size_t)320 << 20));
    s8* qB = (s8*)(ws + ((size_t)352 << 20));
    char* misc = ws + ((size_t)384 << 20);
    s8* wq = (s8*)misc;
    float* part = (float*)(misc + 0x40000);
    float* scaleb = (float*)(misc + 0x90000);
    float* shiftb = (float*)(misc + 0x90400);
    float* pooled = (float*)(misc + 0xA0000);
    s8* qfc = (s8*)(misc + 0x1A0000);
    float* fcbuf = (float*)(misc + 0x1E0000);

    const size_t W_L1B1C1 = 0, W_L1B1C2 = 9216, W_L1B2C1 = 18432, W_L1B2C2 = 27648;
    const size_t W_L2B1C1 = 36864, W_L2B1C2 = 55296, W_L2B1DS = 92160;
    const size_t W_L2B2C1 = 94208, W_L2B2C2 = 131072, W_FC = 167936;

    const long T1 = 1024L * 32 * 32 * 32;
    const long T2 = 1024L * 64 * 16 * 16;

    auto run_stats = [&](const float* x, long total, int C, int Cvalid,
                         const float* g, const float* b, float cnt) {
        int nb = (int)((total + STAT_CHUNK - 1) / STAT_CHUNK);
        k_stats_part<<<nb, 256, 0, stream>>>(x, part, total, C);
        k_stats_final<<<1, 256, 0, stream>>>(part, nb, C, Cvalid, g, b, scaleb, shiftb,
                                             1.0f / cnt);
    };
    auto run_bnact = [&](const float* x, float* fo, s8* qo, long total, int C, int ht) {
        k_bnact<<<nblocks(total), 256, 0, stream>>>(x, scaleb, shiftb, fo, qo, total,
                                                    C - 1, ht);
    };

    k_binw<<<nblocks(9216), 256, 0, stream>>>(F(4), wq + W_L1B1C1, 32, 32, 9);
    k_binw<<<nblocks(9216), 256, 0, stream>>>(F(7), wq + W_L1B1C2, 32, 32, 9);
    k_binw<<<nblocks(9216), 256, 0, stream>>>(F(10), wq + W_L1B2C1, 32, 32, 9);
    k_binw<<<nblocks(9216), 256, 0, stream>>>(F(13), wq + W_L1B2C2, 32, 32, 9);
    k_binw<<<nblocks(18432), 256, 0, stream>>>(F(16), wq + W_L2B1C1, 64, 32, 9);
    k_binw<<<nblocks(36864), 256, 0, stream>>>(F(19), wq + W_L2B1C2, 64, 64, 9);
    k_binw<<<nblocks(2048), 256, 0, stream>>>(F(22), wq + W_L2B1DS, 64, 32, 1);
    k_binw<<<nblocks(36864), 256, 0, stream>>>(F(25), wq + W_L2B2C1, 64, 64, 9);
    k_binw<<<nblocks(36864), 256, 0, stream>>>(F(28), wq + W_L2B2C2, 64, 64, 9);
    k_zero8<<<nblocks(4096), 256, 0, stream>>>(wq + W_FC, 4096);
    k_binw<<<nblocks(2560), 256, 0, stream>>>(F(33), wq + W_FC, 10, 256, 1);

    k_conv1<<<nblocks(1024L * 1024), 256, 0, stream>>>(X, F(1), bufA, 1024);
    run_stats(bufA, T1, 32, 32, F(2), F(3), 1048576.f);
    run_bnact(bufA, bufB, qA, T1, 32, 1);

    k_bconv<32, 2, 3, 1, 1><<<65536, 32, 0, stream>>>(qA, wq + W_L1B1C1, bufA, 1024, 32, 32, 32, 32);
    run_stats(bufA, T1, 32, 32, F(5), F(6), 1048576.f);
    run_bnact(bufA, nullptr, qB, T1, 32, 1);
    k_bconv<32, 2, 3, 1, 1><<<65536, 32, 0, stream>>>(qB, wq + W_L1B1C2, bufA, 1024, 32, 32, 32, 32);
    k_add<<<nblocks(T1), 256, 0, stream>>>(bufA, bufB, T1);
    run_stats(bufA, T1, 32, 32, F(8), F(9), 1048576.f);
    run_bnact(bufA, bufB, qA, T1, 32, 1);

    k_bconv<32, 2, 3, 1, 1><<<65536, 32, 0, stream>>>(qA, wq + W_L1B2C1, bufA, 1024, 32, 32, 32, 32);
    run_stats(bufA, T1, 32, 32, F(11), F(12), 1048576.f);
    run_bnact(bufA, nullptr, qB, T1, 32, 1);
    k_bconv<32, 2, 3, 1, 1><<<65536, 32, 0, stream>>>(qB, wq + W_L1B2C2, bufA, 1024, 32, 32, 32, 32);
    k_add<<<nblocks(T1), 256, 0, stream>>>(bufA, bufB, T1);
    run_stats(bufA, T1, 32, 32, F(14), F(15), 1048576.f);
    run_bnact(bufA, bufB, qA, T1, 32, 1);

    k_bconv<32, 4, 3, 2, 1><<<16384, 32, 0, stream>>>(qA, wq + W_L2B1C1, bufA, 1024, 32, 32, 16, 16);
    run_stats(bufA, T2, 64, 64, F(17), F(18), 262144.f);
    run_bnact(bufA, nullptr, qB, T2, 64, 1);
    k_bconv<64, 4, 3, 1, 1><<<16384, 32, 0, stream>>>(qB, wq + W_L2B1C2, bufA, 1024, 16, 16, 16, 16);
    k_bconv<32, 4, 1, 2, 0><<<16384, 32, 0, stream>>>(qA, wq + W_L2B1DS, bufC, 1024, 32, 32, 16, 16);
    run_stats(bufC, T2, 64, 64, F(23), F(24), 262144.f);
    run_bnact(bufC, bufC, nullptr, T2, 64, 0);
    k_add<<<nblocks(T2), 256, 0, stream>>>(bufA, bufC, T2);
    run_stats(bufA, T2, 64, 64, F(20), F(21), 262144.f);
    run_bnact(bufA, bufB, qA, T2, 64, 1);

    k_bconv<64, 4, 3, 1, 1><<<16384, 32, 0, stream>>>(qA, wq + W_L2B2C1, bufA, 1024, 16, 16, 16, 16);
    run_stats(bufA, T2, 64, 64, F(26), F(27), 262144.f);
    run_bnact(bufA, nullptr, qB, T2, 64, 1);
    k_bconv<64, 4, 3, 1, 1><<<16384, 32, 0, stream>>>(qB, wq + W_L2B2C2, bufA, 1024, 16, 16, 16, 16);
    k_add<<<nblocks(T2), 256, 0, stream>>>(bufA, bufB, T2);

    k_avgpool<<<1024, 256, 0, stream>>>(bufA, pooled, 1024);
    run_stats(pooled, 1024L * 256, 256, 256, F(31), F(32), 1024.f);
    run_bnact(pooled, nullptr, qfc, 1024L * 256, 256, 1);
    k_fc<<<64, 32, 0, stream>>>(qfc, wq + W_FC, F(34), fcbuf);
    run_stats(fcbuf, 1024L * 16, 16, 10, F(35), F(36), 1024.f);
    k_head<<<4, 256, 0, stream>>>(fcbuf, scaleb, shiftb, (float*)d_out, 1024);
}
